// InteractionBlock_60052232732746
// MI455X (gfx1250) — matrix operand order repaired and checked by reading, not compiled, not run on hardware
//
#include <hip/hip_runtime.h>
#include <hip/hip_bf16.h>
#include <stdint.h>

// ---------------- problem constants ----------------
#define N_NODES 10000
#define N_EDGES 60000
#define HCH     64
#define NGAUSS  50
#define KTOT    4160          // 4096 (h x xj outer) + 64 (b3 bias rows)
#define NCHUNK  130           // KTOT / 32
#define NTILES_E (N_EDGES/16) // 3750
#define NJOBS    (NTILES_E/2) // 1875 : one wave = 2 edge tiles (32 edges)
#define NGROUP   32           // groups of 4 chunks (chunks 0..127); +1 bias group

typedef __attribute__((ext_vector_type(16))) __bf16 v16bf;
typedef __attribute__((ext_vector_type(8)))  __bf16 v8bf;
typedef __attribute__((ext_vector_type(8)))  float  v8f;
typedef unsigned int u32x4 __attribute__((ext_vector_type(4)));
typedef int          i32x8 __attribute__((ext_vector_type(8)));
typedef int          i32x4 __attribute__((ext_vector_type(4)));

union XU { v8bf v; uint32_t u[4]; };
union AU { v16bf v; uint32_t u[8]; v8bf h[2]; };
union BU { v16bf v; v8bf h[2]; };

__device__ __forceinline__ float ssp(float v) {
    if (v > 15.0f) return v - 0.6931472f;     // softplus(v) - log(2)
    return __logf(1.0f + __expf(v)) - 0.6931472f;
}

// ---------------- kernel 1: zero agg ----------------
__global__ void k_zero(float* p, int n) {
    int t = blockIdx.x * 256 + threadIdx.x;
    if (t < n) p[t] = 0.0f;
}

// ---------------- kernel 2: pack W3 (+b3) into WMMA-B tile layout, bf16 ----
__global__ void k_pack_w3(const float* __restrict__ W3, const float* __restrict__ b3,
                          __bf16* __restrict__ w3p) {
    int p = blockIdx.x * 256 + threadIdx.x;
    if (p >= KTOT * HCH) return;
    int tile = p >> 9, w = p & 511;
    int lane = w >> 4, idx = w & 15;
    int v = idx >> 1, par = idx & 1;
    int c = tile >> 2, t = tile & 3;
    int np = lane & 15, half = lane >> 4;
    int i  = t * 16 + np;
    int kk = c * 32 + half * 16 + 2 * v + par;
    float val;
    if (kk < 4096) {
        int f = kk >> 6, j = kk & 63;
        val = W3[(size_t)f * 4096 + i * 64 + j];
    } else {
        int j = kk - 4096;
        val = b3[i * 64 + j];
    }
    w3p[p] = (__bf16)val;
}

// ---------------- kernel 3: rbf -> ssp(rbf@W1+b1), fp32 ----------------
__global__ void k_h1(const float* __restrict__ ea, const float* __restrict__ W1,
                     const float* __restrict__ b1, float* __restrict__ h1) {
    int tid = blockIdx.x * 256 + threadIdx.x;
    if (tid >= N_EDGES * HCH) return;
    int e = tid >> 6, i = tid & 63;
    float d = ea[e];
    const float step  = 5.0f / 49.0f;
    const float coeff = -0.5f / (step * step);
    float acc = b1[i];
    #pragma unroll 5
    for (int g = 0; g < NGAUSS; ++g) {
        float dd = d - (float)g * step;
        acc += __expf(coeff * dd * dd) * W1[g * 64 + i];
    }
    h1[tid] = ssp(acc);
}

// ------- kernel 4: h = ssp(h1@W2+b2) -> bf16 ; gather xj = x[src] -> bf16 ----
__global__ __launch_bounds__(256) void k_h2_gather(
    const float* __restrict__ h1, const float* __restrict__ W2,
    const float* __restrict__ b2, const float* __restrict__ x,
    const long long* __restrict__ eidx, __bf16* __restrict__ hb,
    __bf16* __restrict__ xj) {
    __shared__ float s[4][64];
    int le = threadIdx.x >> 6, i = threadIdx.x & 63;
    int e = blockIdx.x * 4 + le;
    if (e < N_EDGES) s[le][i] = h1[(size_t)e * 64 + i];
    __syncthreads();
    if (e >= N_EDGES) return;
    float acc = b2[i];
    #pragma unroll 8
    for (int f = 0; f < 64; ++f) acc += s[le][f] * W2[f * 64 + i];
    hb[(size_t)e * 64 + i] = (__bf16)ssp(acc);
    long long srce = eidx[e];
    xj[(size_t)e * 64 + i] = (__bf16)x[srce * 64 + i];
}

// ---- TDM: issue a 1-D tensor_load_to_lds (nwords8 x 8B), D# per ISA 8.3/8.4 ----
__device__ __forceinline__ void tdm_issue(const void* gsrc, uint32_t lds_off,
                                          uint32_t nwords8) {
    uint64_t ga = (uint64_t)(uintptr_t)gsrc;
    u32x4 g0;
    g0[0] = 1u;                                          // count=1, user mode
    g0[1] = lds_off;                                     // lds_addr (bytes)
    g0[2] = (uint32_t)ga;                                // global_addr[31:0]
    g0[3] = (uint32_t)((ga >> 32) & 0x01FFFFFFu) | (2u << 30); // addr[56:32]|type=2
    i32x8 g1;
    g1[0] = (int)(3u << 16);                             // data_size=8B, mask=0
    g1[1] = (int)((nwords8 & 0xffffu) << 16);            // tensor_dim0[15:0]
    g1[2] = (int)((nwords8 >> 16) | (1u << 16));         // tensor_dim0[31:16] | tensor_dim1=1
    g1[3] = (int)(nwords8 << 16);                        // tile_dim0 = nwords8
    g1[4] = 0;                                           // tile_dim1=0 (1-D), tile_dim2=0
    g1[5] = (int)nwords8;                                // tensor_dim0_stride lo
    g1[6] = 0;
    g1[7] = 0;
    i32x4 z4 = {0, 0, 0, 0};                             // groups 2/3 unused (<=2D)
    i32x8 z8 = {0, 0, 0, 0, 0, 0, 0, 0};
    __builtin_amdgcn_tensor_load_to_lds(g0, g1, z4, z4, z8, 0);
}

// ---------------- kernel 5: fused filter-GEMM + message + scatter ----------
// Block = 8 waves; wave0 TDM-stages 16KB B-groups (4 chunks) into LDS double
// buffer, all waves consume via ds_load_b128. One wave = 2 edge tiles.
#define WMMA4(ACC_, A_)                                                   \
    _Pragma("unroll")                                                     \
    for (int t_ = 0; t_ < 4; ++t_)                                        \
        ACC_[t_] = __builtin_amdgcn_wmma_f32_16x16x32_bf16(               \
            false, (A_).v, false, Bt[t_].v, (short)0, ACC_[t_],           \
            false, false)

__global__ __launch_bounds__(256) void k_msg_wmma(
    const __bf16* __restrict__ hbuf, const __bf16* __restrict__ xjbuf,
    const __bf16* __restrict__ w3p, const long long* __restrict__ eidx,
    float* __restrict__ agg) {
    __shared__ __align__(16) unsigned char blds[2][16384];   // 32 KB B stage
    const int wave = threadIdx.x >> 5;
    const int lane = threadIdx.x & 31;
    const int job  = blockIdx.x * 8 + wave;
    const bool active = (job < NJOBS);
    const int jobc = active ? job : 0;        // idle waves shadow job 0 (no scatter)
    const int e0   = jobc * 32;
    const int m    = lane & 15;
    const int half = lane >> 4;

    // ---- preload xj A-layout runs for both tiles ----
    XU X0[4], X1[4];
    const __bf16* xr0 = xjbuf + (size_t)(e0 + m) * 64;
    const __bf16* xr1 = xjbuf + (size_t)(e0 + 16 + m) * 64;
    #pragma unroll
    for (int q = 0; q < 4; ++q) {
        X0[q].v = *(const v8bf*)(xr0 + (q >> 1) * 32 + half * 16 + (q & 1) * 8);
        X1[q].v = *(const v8bf*)(xr1 + (q >> 1) * 32 + half * 16 + (q & 1) * 8);
    }
    const uint32_t* hg0 = (const uint32_t*)(hbuf + (size_t)(e0 + m) * 64);
    const uint32_t* hg1 = (const uint32_t*)(hbuf + (size_t)(e0 + 16 + m) * 64);

    const v8f vz = {0.f,0.f,0.f,0.f,0.f,0.f,0.f,0.f};
    v8f acc0[4], acc1[4];
    #pragma unroll
    for (int t = 0; t < 4; ++t) { acc0[t] = vz; acc1[t] = vz; }

    // ---- prologue: stage group 0 ----
    if (wave == 0) {
        tdm_issue(w3p, (uint32_t)(uintptr_t)(void*)&blds[0][0], 2048);
        __builtin_amdgcn_s_wait_tensorcnt(0);
    }
    __syncthreads();

    #pragma unroll 1
    for (int g = 0; g < NGROUP; ++g) {
        if (wave == 0) {               // stage next group (g+1); group 32 = 8KB
            tdm_issue(w3p + (size_t)(g + 1) * 8192,
                      (uint32_t)(uintptr_t)(void*)&blds[(g + 1) & 1][0],
                      (g + 1 == NGROUP) ? 1024u : 2048u);
        }
        const uint32_t hp0 = hg0[g];   // h[m, 2g] | h[m, 2g+1]<<16
        const uint32_t hp1 = hg1[g];
        const unsigned char* lb = &blds[g & 1][0];
        #pragma unroll
        for (int sub = 0; sub < 4; ++sub) {
            const unsigned char* cb = lb + sub * 4096 + lane * 32;
            BU Bt[4];
            #pragma unroll
            for (int t = 0; t < 4; ++t) {
                Bt[t].h[0] = *(const v8bf*)(cb + t * 1024);
                Bt[t].h[1] = *(const v8bf*)(cb + t * 1024 + 16);
            }
            const uint32_t hu0 = (sub < 2) ? (hp0 & 0xffffu) : (hp0 >> 16);
            const uint32_t hu1 = (sub < 2) ? (hp1 & 0xffffu) : (hp1 >> 16);
            const uint32_t hs0 = hu0 * 0x10001u;
            const uint32_t hs1 = hu1 * 0x10001u;
            const int js = (sub & 1) * 2;
            AU a0, a1;                 // u = h[m,f] * xj[m,j]
            #pragma unroll
            for (int p = 0; p < 4; ++p) {
                asm("v_pk_mul_bf16 %0, %1, %2" : "=v"(a0.u[p])     : "v"(hs0), "v"(X0[js].u[p]));
                asm("v_pk_mul_bf16 %0, %1, %2" : "=v"(a0.u[p + 4]) : "v"(hs0), "v"(X0[js + 1].u[p]));
                asm("v_pk_mul_bf16 %0, %1, %2" : "=v"(a1.u[p])     : "v"(hs1), "v"(X1[js].u[p]));
                asm("v_pk_mul_bf16 %0, %1, %2" : "=v"(a1.u[p + 4]) : "v"(hs1), "v"(X1[js + 1].u[p]));
            }
            WMMA4(acc0, a0);
            WMMA4(acc1, a1);
        }
        if (wave == 0) __builtin_amdgcn_s_wait_tensorcnt(0);
        __syncthreads();               // buffer (g+1)&1 ready; g&1 free to reuse
    }

    // ---- bias group (chunks 128,129 in blds[0]): A = xj directly ----
    #pragma unroll
    for (int sub = 0; sub < 2; ++sub) {
        const unsigned char* cb = &blds[0][0] + sub * 4096 + lane * 32;
        BU Bt[4];
        #pragma unroll
        for (int t = 0; t < 4; ++t) {
            Bt[t].h[0] = *(const v8bf*)(cb + t * 1024);
            Bt[t].h[1] = *(const v8bf*)(cb + t * 1024 + 16);
        }
        const int js = sub * 2;
        AU a0, a1;
        a0.h[0] = X0[js].v; a0.h[1] = X0[js + 1].v;
        a1.h[0] = X1[js].v; a1.h[1] = X1[js + 1].v;
        WMMA4(acc0, a0);
        WMMA4(acc1, a1);
    }

    // ---- scatter-add (masked for idle shadow waves) ----
    if (active) {
        #pragma unroll
        for (int r = 0; r < 8; ++r) {
            const int er = r + half * 8;
            {
                int e = e0 + er;
                int d = (int)eidx[N_EDGES + e];
                float* row = agg + (size_t)d * 64 + (lane & 15);
                #pragma unroll
                for (int t = 0; t < 4; ++t)
                    unsafeAtomicAdd(row + t * 16, acc0[t][r]);
            }
            {
                int e = e0 + 16 + er;
                int d = (int)eidx[N_EDGES + e];
                float* row = agg + (size_t)d * 64 + (lane & 15);
                #pragma unroll
                for (int t = 0; t < 4; ++t)
                    unsafeAtomicAdd(row + t * 16, acc1[t][r]);
            }
        }
    }
}

// ---------------- kernel 6: node FFN + residual ----------------
__global__ __launch_bounds__(64) void k_ffn(
    const float* __restrict__ agg, const float* __restrict__ Wa1,
    const float* __restrict__ ba1, const float* __restrict__ Wa2,
    const float* __restrict__ ba2, const float* __restrict__ x,
    float* __restrict__ out) {
    __shared__ float a[64], t1[64];
    int n = blockIdx.x, i = threadIdx.x;
    a[i] = agg[(size_t)n * 64 + i];
    __syncthreads();
    float s = ba1[i];
    #pragma unroll 8
    for (int f = 0; f < 64; ++f) s += a[f] * Wa1[f * 64 + i];
    t1[i] = s / (1.0f + __expf(-s));
    __syncthreads();
    float o = ba2[i];
    #pragma unroll 8
    for (int f = 0; f < 64; ++f) o += t1[f] * Wa2[f * 64 + i];
    out[(size_t)n * 64 + i] = o + x[(size_t)n * 64 + i];
}

// ---------------- workspace layout ----------------
#define O_W3P 0u
#define O_H1  532480u
#define O_HB  (O_H1 + 15360000u)
#define O_XJ  (O_HB + 7680000u)
#define O_AGG (O_XJ + 7680000u)

extern "C" void kernel_launch(void* const* d_in, const int* in_sizes, int n_in,
                              void* d_out, int out_size, void* d_ws, size_t ws_size,
                              hipStream_t stream) {
    const float*     x    = (const float*)d_in[0];
    const long long* eidx = (const long long*)d_in[1];
    const float*     ea   = (const float*)d_in[2];
    const float*     W1   = (const float*)d_in[3];
    const float*     b1   = (const float*)d_in[4];
    const float*     W2   = (const float*)d_in[5];
    const float*     b2   = (const float*)d_in[6];
    const float*     W3   = (const float*)d_in[7];
    const float*     b3   = (const float*)d_in[8];
    const float*     Wa1  = (const float*)d_in[9];
    const float*     ba1  = (const float*)d_in[10];
    const float*     Wa2  = (const float*)d_in[11];
    const float*     ba2  = (const float*)d_in[12];
    float* out = (float*)d_out;

    char* ws = (char*)d_ws;
    __bf16* w3p = (__bf16*)(ws + O_W3P);
    float*  h1  = (float*) (ws + O_H1);
    __bf16* hb  = (__bf16*)(ws + O_HB);
    __bf16* xj  = (__bf16*)(ws + O_XJ);
    float*  agg = (float*) (ws + O_AGG);

    k_zero<<<(N_NODES * HCH + 255) / 256, 256, 0, stream>>>(agg, N_NODES * HCH);
    k_pack_w3<<<(KTOT * HCH + 255) / 256, 256, 0, stream>>>(W3, b3, w3p);
    k_h1<<<(N_EDGES * HCH + 255) / 256, 256, 0, stream>>>(ea, W1, b1, h1);
    k_h2_gather<<<(N_EDGES + 3) / 4, 256, 0, stream>>>(h1, W2, b2, x, eidx, hb, xj);
    k_msg_wmma<<<(NJOBS + 7) / 8, 256, 0, stream>>>(hb, xj, w3p, eidx, agg);
    k_ffn<<<N_NODES, 64, 0, stream>>>(agg, Wa1, ba1, Wa2, ba2, x, out);
}
